// VectorQuantizationLayer_20718922236144
// MI455X (gfx1250) — matrix operand order repaired and checked by reading, not compiled, not run on hardware
//
#include <hip/hip_runtime.h>
#include <hip/hip_bf16.h>
#include <hip/hip_fp16.h>

// VQ-VAE vector quantization for MI455X (gfx1250, wave32, WMMA).
//
// Shapes: x [16,256,64,64] f32, weight [2048,256] f32.
// out = concat( quantized [16,256,64,64], loss, perplexity ), all f32.
//
// Distances argmin via f16 WMMA (16x16x32, fp32 accum). Weight pre-scaled by
// 1024 so |w| <= 1/2048 sits in f16 normal range. |x|^2 dropped (row-constant).
// v2: 32 rows per wave (2 independent WMMA accumulator chains, 2x B reuse) and
// double-buffered async global->LDS B-tile copies (ASYNCcnt) to hide fetch
// latency behind the WMMA stream.

typedef __attribute__((ext_vector_type(16))) _Float16 v16h;
typedef __attribute__((ext_vector_type(8)))  float    v8f;

#define NVEC   2048      // codebook entries K
#define VD     256       // vector dim D
#define BATCH  16
#define HWD    4096      // 64*64
#define NROWS  65536     // BATCH * HWD
#define WSCALE 1024.0f
#define NCTILE (NVEC / 16)   // 128 column tiles
#define NRTILE (NROWS / 16)  // 4096 row tiles
#define TILEHALFS (8 * 32 * 16)   // halves per 16-col B tile (8 KB)

// Async global->LDS 16-byte copy (gfx1250; tracked by ASYNCcnt).
__device__ __forceinline__ void async_copy_b128(unsigned ldsOff, const void* gptr) {
    asm volatile("global_load_async_to_lds_b128 %0, %1, off"
                 :: "v"(ldsOff), "v"(gptr) : "memory");
}

// ---------------------------------------------------------------------------
// Kernel 0: zero the histogram and loss accumulator (ws is poisoned, not zeroed)
__global__ __launch_bounds__(256) void vq_init_kernel(unsigned* hist, float* lossAcc) {
    int i = blockIdx.x * 256 + threadIdx.x;
    if (i < NVEC) hist[i] = 0u;
    if (i == 0)   lossAcc[0] = 0.0f;
}

// ---------------------------------------------------------------------------
// Kernel 1: pack weight into f16 WMMA B-matrix lane layout, scaled by WSCALE.
// B 32x16 f16 layout used here (contiguous K, as in A): lanes 0-15 -> col N=lane, K=0..15 (K=2v+p);
// lanes 16-31 -> col N=lane-16, K=16..31.
__global__ __launch_bounds__(256) void vq_packB_kernel(const float* __restrict__ w,
                                                       _Float16* __restrict__ packedB) {
    const int ct   = blockIdx.x;          // column tile 0..127
    const int tid  = threadIdx.x;
    const int dc   = tid >> 5;            // d-chunk 0..7 (32 K-values each)
    const int lane = tid & 31;
    const int col  = ct * 16 + (lane & 15);
    const int hf   = lane >> 4;

    union { _Float16 h[16]; float4 f4[2]; } tmp;
#pragma unroll
    for (int s = 0; s < 16; ++s) {
        const int v = s >> 1, p = s & 1;
        const int kk = hf * 16 + 2 * v + p;      // B layout
        const int d  = dc * 32 + kk;
        tmp.h[s] = (_Float16)(w[(size_t)col * VD + d] * WSCALE);
    }
    float4* dst = (float4*)(packedB + (((size_t)ct * 8 + dc) * 32 + lane) * 16);
    dst[0] = tmp.f4[0];
    dst[1] = tmp.f4[1];
}

// ---------------------------------------------------------------------------
// Kernel 2: wnorm[k] = |e_k|^2 (fp32, unscaled)
__global__ __launch_bounds__(256) void vq_wnorm_kernel(const float* __restrict__ w,
                                                       float* __restrict__ wnorm) {
    const int k = blockIdx.x * 256 + threadIdx.x;
    if (k >= NVEC) return;
    float s = 0.0f;
    for (int d = 0; d < VD; ++d) {
        const float v = w[(size_t)k * VD + d];
        s += v * v;
    }
    wnorm[k] = s;
}

// ---------------------------------------------------------------------------
// Kernel 3: WMMA distance sweep + running argmin.
// One wave owns a 32-row block (2 row tiles, 2 acc chains); 8 waves/block
// share double-buffered B tiles loaded with async global->LDS DMA.
__global__ __launch_bounds__(256) void vq_argmin_kernel(const float* __restrict__ x,
                                                        const _Float16* __restrict__ packedB,
                                                        const float* __restrict__ wnorm,
                                                        int* __restrict__ idxOut) {
    __shared__ alignas(32) _Float16 ldsB[2][TILEHALFS];   // 2 x 8 KB

    const int tid     = threadIdx.x;
    const int lane    = tid & 31;
    const int wave    = tid >> 5;
    const int rowBase = (blockIdx.x * 8 + wave) * 32;     // 32 rows per wave

    // --- Build A fragments (f16 16x32 per d-chunk) directly from x ---------
    // A layout: lane l row M=l&15; half h=l>>4;
    // element s: K = 16h+s (matches packed B)
    const int M  = lane & 15;
    const int hf = lane >> 4;

    v16h a[2][8];
#pragma unroll
    for (int m = 0; m < 2; ++m) {
        const int n  = rowBase + m * 16 + M;   // flat row = b*4096 + (h*64+w)
        const int b  = n >> 12;
        const int hw = n & 4095;
        const float* xb = x + (size_t)b * (VD * HWD) + hw;
#pragma unroll
        for (int dc = 0; dc < 8; ++dc) {
#pragma unroll
            for (int s = 0; s < 16; ++s) {
                const int v = s >> 1, p = s & 1;
                const int kk = hf * 16 + 2 * v + p;
                const int d  = dc * 32 + kk;
                a[m][dc][s] = (_Float16)xb[(size_t)d * HWD];
            }
        }
    }

    float best[2][8];
    int   bidx[2][8];
#pragma unroll
    for (int m = 0; m < 2; ++m)
#pragma unroll
        for (int r = 0; r < 8; ++r) { best[m][r] = 3.402823466e38f; bidx[m][r] = 0; }

    // Prologue: async-load tile 0 into buffer 0 (32 B per thread).
    {
        const _Float16* src = packedB + (size_t)tid * 16;
        const unsigned  dst = (unsigned)(uintptr_t)&ldsB[0][tid * 16];
        async_copy_b128(dst,      src);
        async_copy_b128(dst + 16, (const char*)src + 16);
    }

    for (int ct = 0; ct < NCTILE; ++ct) {
        const int buf = ct & 1;
        // Prefetch next tile into the other buffer, then wait for current tile.
        if (ct + 1 < NCTILE) {
            const _Float16* src = packedB + (size_t)(ct + 1) * TILEHALFS + (size_t)tid * 16;
            const unsigned  dst = (unsigned)(uintptr_t)&ldsB[buf ^ 1][tid * 16];
            async_copy_b128(dst,      src);
            async_copy_b128(dst + 16, (const char*)src + 16);
            asm volatile("s_wait_asynccnt 0x2" ::: "memory");
        } else {
            asm volatile("s_wait_asynccnt 0x0" ::: "memory");
        }
        const int   ncol = ct * 16 + (lane & 15);
        const float wn   = wnorm[ncol];
        __syncthreads();   // current tile fully in LDS, visible to all waves

        v8f acc0 = {}, acc1 = {};
#pragma unroll
        for (int dc = 0; dc < 8; ++dc) {
            v16h bfrag = *(const v16h*)(&ldsB[buf][((size_t)dc * 32 + lane) * 16]);
            acc0 = __builtin_amdgcn_wmma_f32_16x16x32_f16(
                       false, a[0][dc], false, bfrag, (short)0, acc0, false, false);
            acc1 = __builtin_amdgcn_wmma_f32_16x16x32_f16(
                       false, a[1][dc], false, bfrag, (short)0, acc1, false, false);
        }

        // dist = |e|^2 - 2*x.e  (sim scaled by WSCALE); keep first min per lane
#pragma unroll
        for (int r = 0; r < 8; ++r) {
            const float d0 = wn - (2.0f / WSCALE) * acc0[r];
            if (d0 < best[0][r]) { best[0][r] = d0; bidx[0][r] = ncol; }
            const float d1 = wn - (2.0f / WSCALE) * acc1[r];
            if (d1 < best[1][r]) { best[1][r] = d1; bidx[1][r] = ncol; }
        }
        __syncthreads();   // all waves done with this buffer before it is reloaded
    }

    // cross-lane argmin within each 16-lane group (D layout: N = lane&15)
#pragma unroll
    for (int m = 0; m < 2; ++m)
#pragma unroll
        for (int r = 0; r < 8; ++r) {
#pragma unroll
            for (int msk = 1; msk < 16; msk <<= 1) {
                const float ov = __shfl_xor(best[m][r], msk, 32);
                const int   oi = __shfl_xor(bidx[m][r], msk, 32);
                if (ov < best[m][r] || (ov == best[m][r] && oi < bidx[m][r])) {
                    best[m][r] = ov; bidx[m][r] = oi;
                }
            }
        }

    // D layout: VGPR r holds M=r (lanes 0-15) and M=8+r (lanes 16-31)
    if ((lane & 15) == 0) {
#pragma unroll
        for (int m = 0; m < 2; ++m) {
            const int mbase = rowBase + m * 16 + hf * 8;
#pragma unroll
            for (int r = 0; r < 8; ++r) idxOut[mbase + r] = bidx[m][r];
        }
    }
}

// ---------------------------------------------------------------------------
// Kernel 4: gather codebook rows -> out [B,C,H,W] (coalesced over hw),
// accumulate loss sum and codebook usage histogram.
__global__ __launch_bounds__(256) void vq_gather_kernel(const float* __restrict__ x,
                                                        const float* __restrict__ w,
                                                        const int* __restrict__ idx,
                                                        float* __restrict__ out,
                                                        float* __restrict__ lossAcc,
                                                        unsigned* __restrict__ hist) {
    __shared__ float red[256];
    const size_t gid = (size_t)blockIdx.x * 256 + threadIdx.x;
    const int d = (int)(gid >> 16);          // 0..255
    const int n = (int)(gid & 65535);        // 0..65535, contiguous across threads
    const int b = n >> 12;
    const int hw = n & 4095;
    const size_t xo = (size_t)b * (VD * HWD) + (size_t)d * HWD + hw;

    const int   k  = idx[n];
    const float q  = w[(size_t)k * VD + d];  // scattered but L2-resident (2 MB)
    const float xv = x[xo];
    out[xo] = q;                             // straight-through: out == quantized

    if (d == 0) atomicAdd(&hist[k], 1u);

    const float diff = q - xv;
    red[threadIdx.x] = diff * diff;
    __syncthreads();
    for (int s = 128; s > 0; s >>= 1) {
        if (threadIdx.x < s) red[threadIdx.x] += red[threadIdx.x + s];
        __syncthreads();
    }
    if (threadIdx.x == 0) atomicAdd(lossAcc, red[0]);
}

// ---------------------------------------------------------------------------
// Kernel 5: finalize scalars: loss = 1.25 * mean((q-x)^2), perplexity.
__global__ __launch_bounds__(256) void vq_final_kernel(const float* __restrict__ lossAcc,
                                                       const unsigned* __restrict__ hist,
                                                       float* __restrict__ scalars) {
    __shared__ float red[256];
    float s = 0.0f;
    for (int k = threadIdx.x; k < NVEC; k += 256) {
        const float p = (float)hist[k] * (1.0f / (float)NROWS);
        s += p * logf(p + 1e-10f);
    }
    red[threadIdx.x] = s;
    __syncthreads();
    for (int st = 128; st > 0; st >>= 1) {
        if (threadIdx.x < st) red[threadIdx.x] += red[threadIdx.x + st];
        __syncthreads();
    }
    if (threadIdx.x == 0) {
        scalars[0] = 1.25f * lossAcc[0] / (float)((size_t)NROWS * VD);  // loss
        scalars[1] = expf(-red[0]);                                     // perplexity
    }
}

// ---------------------------------------------------------------------------
extern "C" void kernel_launch(void* const* d_in, const int* in_sizes, int n_in,
                              void* d_out, int out_size, void* d_ws, size_t ws_size,
                              hipStream_t stream) {
    const float* x = (const float*)d_in[0];   // [16,256,64,64]
    const float* w = (const float*)d_in[1];   // [2048,256]
    float* out = (float*)d_out;               // [16,256,64,64] ++ [loss, perplexity]

    // Scratch layout (~1.3 MB total)
    char* ws = (char*)d_ws;
    _Float16* packedB = (_Float16*)(ws);                       // 128*8*32*16*2 = 1 MB
    size_t off = (size_t)NCTILE * TILEHALFS * sizeof(_Float16);
    float*    wnorm   = (float*)(ws + off);    off += NVEC * sizeof(float);
    int*      idx     = (int*)(ws + off);      off += (size_t)NROWS * sizeof(int);
    unsigned* hist    = (unsigned*)(ws + off); off += NVEC * sizeof(unsigned);
    float*    lossAcc = (float*)(ws + off);

    float* scalars = out + (size_t)BATCH * VD * HWD;   // out elements, then scalars

    vq_init_kernel  <<<(NVEC + 255) / 256, 256, 0, stream>>>(hist, lossAcc);
    vq_packB_kernel <<<NCTILE, 256, 0, stream>>>(w, packedB);
    vq_wnorm_kernel <<<(NVEC + 255) / 256, 256, 0, stream>>>(w, wnorm);
    vq_argmin_kernel<<<NRTILE / 16, 256, 0, stream>>>(x, packedB, wnorm, idx);
    vq_gather_kernel<<<((size_t)NROWS * VD) / 256, 256, 0, stream>>>(x, w, idx, out,
                                                                     lossAcc, hist);
    vq_final_kernel <<<1, 256, 0, stream>>>(lossAcc, hist, scalars);
}
